// ResBlockMod_79903571575297
// MI455X (gfx1250) — matrix operand order repaired and checked by reading, not compiled, not run on hardware
//
#include <hip/hip_runtime.h>
#include <hip/hip_bf16.h>

typedef __attribute__((ext_vector_type(16))) __bf16 v16bf;
typedef __attribute__((ext_vector_type(8)))  __bf16 v8bf;
typedef __attribute__((ext_vector_type(8)))  float  v8f;

constexpr int NB   = 32;    // batch
constexpr int ICH  = 128;   // in channels
constexpr int OCH  = 256;   // out channels
constexpr int HH   = 64;
constexpr int WW   = 64;
constexpr int NG   = 32;    // groups
constexpr int HW   = HH * WW;

// ---------------------------------------------------------------------------
// CDNA5 async global->LDS copy (ASYNCcnt-tracked, no VGPR data movement).
// dst: LDS byte offset (per-lane VGPR), src: 16B-aligned global address.
// ---------------------------------------------------------------------------
__device__ __forceinline__ void async_ld_b128(unsigned dst, const void* src) {
#if defined(__HIP_DEVICE_COMPILE__)
  asm volatile("global_load_async_to_lds_b128 %0, %1, off"
               :: "v"(dst), "v"((unsigned long long)src) : "memory");
#endif
}
__device__ __forceinline__ void wait_async0() {
#if defined(__HIP_DEVICE_COMPILE__)
  asm volatile("s_wait_asynccnt 0x0" ::: "memory");
#endif
}

// ---------------------------------------------------------------------------
// GroupNorm statistics: one block per (b, g), mean + rstd
// ---------------------------------------------------------------------------
template<int C>
__global__ __launch_bounds__(256)
void gn_stats_kernel(const float* __restrict__ x, float* __restrict__ stats) {
  const int bg = blockIdx.x;
  const int b = bg / NG, g = bg % NG;
  constexpr int CPG = C / NG;
  const long long base = ((long long)b * C + (long long)g * CPG) * HW;
  const int n = CPG * HW;
  float s = 0.f, s2 = 0.f;
  for (int i = threadIdx.x; i < n; i += 256) {
    float v = x[base + i];
    s += v; s2 += v * v;
  }
  __shared__ float red0[256], red1[256];
  red0[threadIdx.x] = s; red1[threadIdx.x] = s2;
  __syncthreads();
  for (int off = 128; off > 0; off >>= 1) {
    if (threadIdx.x < off) {
      red0[threadIdx.x] += red0[threadIdx.x + off];
      red1[threadIdx.x] += red1[threadIdx.x + off];
    }
    __syncthreads();
  }
  if (threadIdx.x == 0) {
    float mu  = red0[0] / (float)n;
    float var = red1[0] / (float)n - mu * mu;
    stats[bg * 2 + 0] = mu;
    stats[bg * 2 + 1] = rsqrtf(var + 1e-5f);
  }
}

// ---------------------------------------------------------------------------
// GroupNorm normalize + swish -> bf16 activations
// ---------------------------------------------------------------------------
template<int C>
__global__ __launch_bounds__(256)
void gn_swish_bf16_kernel(const float* __restrict__ x, const float* __restrict__ stats,
                          const float* __restrict__ gamma, const float* __restrict__ beta,
                          __bf16* __restrict__ out) {
  constexpr int CPG = C / NG;
  const long long total = (long long)NB * C * HW;
  const long long stride = (long long)gridDim.x * 256;
  for (long long i = (long long)blockIdx.x * 256 + threadIdx.x; i < total; i += stride) {
    int c = (int)((i >> 12) % C);       // HW == 4096
    int b = (int)((i >> 12) / C);
    int g = c / CPG;
    float mu = stats[(b * NG + g) * 2 + 0];
    float rs = stats[(b * NG + g) * 2 + 1];
    float v = (x[i] - mu) * rs * gamma[c] + beta[c];
    float sw = v / (1.f + __expf(-v));
    out[i] = (__bf16)sw;
  }
}

// ---------------------------------------------------------------------------
// fp32 -> bf16 convert (for shortcut input)
// ---------------------------------------------------------------------------
__global__ __launch_bounds__(256)
void cvt_bf16_kernel(const float* __restrict__ in, __bf16* __restrict__ out, long long n) {
  const long long stride = (long long)gridDim.x * 256;
  for (long long i = (long long)blockIdx.x * 256 + threadIdx.x; i < n; i += stride)
    out[i] = (__bf16)in[i];
}

// ---------------------------------------------------------------------------
// Weight swizzle: [O][IC][T] fp32 -> [T][O][IC] bf16  (T = KH*KW taps)
// ---------------------------------------------------------------------------
__global__ __launch_bounds__(256)
void wswizzle_kernel(const float* __restrict__ in, __bf16* __restrict__ out,
                     int O, int IC, int T) {
  const long long n = (long long)O * IC * T;
  const long long stride = (long long)gridDim.x * 256;
  for (long long i = (long long)blockIdx.x * 256 + threadIdx.x; i < n; i += stride) {
    int t  = (int)(i % T);
    int ic = (int)((i / T) % IC);
    int o  = (int)(i / ((long long)T * IC));
    out[((long long)t * O + o) * IC + ic] = (__bf16)in[i];
  }
}

// ---------------------------------------------------------------------------
// S_o = sum_{i,kh,kw} mod_w[o]^2   (one block per output channel)
// ---------------------------------------------------------------------------
__global__ __launch_bounds__(256)
void modw_sumsq_kernel(const float* __restrict__ mod_w, float* __restrict__ S) {
  const int o = blockIdx.x;
  const float* p = mod_w + (long long)o * OCH * 9;
  float s = 0.f;
  for (int i = threadIdx.x; i < OCH * 9; i += 256) { float v = p[i]; s += v * v; }
  __shared__ float red[256];
  red[threadIdx.x] = s;
  __syncthreads();
  for (int off = 128; off > 0; off >>= 1) {
    if (threadIdx.x < off) red[threadIdx.x] += red[threadIdx.x + off];
    __syncthreads();
  }
  if (threadIdx.x == 0) S[o] = red[0];
}

// ---------------------------------------------------------------------------
// temb projection: out[b,o] = swish(temb[b]) . temb_w[o] + temb_b[o]
// ---------------------------------------------------------------------------
__global__ __launch_bounds__(256)
void temb_proj_kernel(const float* __restrict__ temb, const float* __restrict__ w,
                      const float* __restrict__ bias, float* __restrict__ out) {
  int i = blockIdx.x * 256 + threadIdx.x;
  if (i >= NB * OCH) return;
  int b = i / OCH, o = i % OCH;
  float acc = bias[o];
  for (int t = 0; t < 512; ++t) {
    float tv = temb[b * 512 + t];
    tv = tv / (1.f + __expf(-tv));
    acc += tv * w[o * 512 + t];
  }
  out[i] = acc;
}

// ---------------------------------------------------------------------------
// Demod scale: style2 = style . style_w^T + style_b;  s = style2 + 1
// scale[b,o] = s * rsqrt(s*s*S_o + 1e-8)   (conv linearity folds wmod into this)
// ---------------------------------------------------------------------------
__global__ __launch_bounds__(256)
void style_scale_kernel(const float* __restrict__ style, const float* __restrict__ w,
                        const float* __restrict__ bias, const float* __restrict__ S,
                        float* __restrict__ scale) {
  int i = blockIdx.x * 256 + threadIdx.x;
  if (i >= NB * OCH) return;
  int b = i / OCH, o = i % OCH;
  float acc = bias[o];
  for (int t = 0; t < 512; ++t) acc += style[b * 512 + t] * w[o * 512 + t];
  float s = acc + 1.f;
  scale[i] = s * rsqrtf(s * s * S[o] + 1e-8f);
}

// ---------------------------------------------------------------------------
// Implicit-GEMM conv via WMMA bf16.
//   act : [NB][IC][H][W] bf16        wgt : [KD*KD][OCH][IC] bf16 (swizzled)
//   Block = (b, y): all 256 out channels x 64 pixels of row y. 8 waves,
//   wave w owns channels [32w, 32w+32) -> 2 M-tiles x 4 N-tiles = 8 wmma accs.
//   Weights staged one kh-row (KD taps) at a time via async global->LDS b128.
//   MODE 0: out = gemm + bias[m] + pervec[b,m]        (conv1 + temb add)
//   MODE 1: out = gemm + bias[m]                      (1x1 shortcut)
//   MODE 2: out += gemm * pervec[b,m]                 (modulated conv, demod scale)
// ---------------------------------------------------------------------------
template<int IC, int KD, int MODE>
__global__ __launch_bounds__(256)
void conv_wmma_kernel(const __bf16* __restrict__ act, const __bf16* __restrict__ wgt,
                      const float* __restrict__ bias, const float* __restrict__ pervec,
                      float* __restrict__ out) {
  constexpr int PAD = KD / 2;
  constexpr int LW  = WW + 2 * PAD;
  constexpr int KT  = 32;                       // K per WMMA (bf16)

  __shared__ __align__(16) __bf16 actlds[KD][LW][KT];   // K innermost: B-frag = one 32B read
  __shared__ __align__(16) __bf16 wlds[KD][OCH][KT];    // one kh-row of taps (KD=3: 48KB)

  const int b = blockIdx.x / HH;
  const int y = blockIdx.x % HH;
  const int tid   = threadIdx.x;
  const int wave  = tid >> 5;
  const int lane  = tid & 31;
  const int lhalf = lane >> 4;                  // which 16-lane half
  const int l16   = lane & 15;
  const int m_base = wave * 32;

  const unsigned wlds_base = (unsigned)(unsigned long long)(uintptr_t)&wlds[0][0][0];

  v8f acc[2][4] = {};

  for (int ic0 = 0; ic0 < IC; ic0 += KT) {
    __syncthreads();
    // ---- stage activation slab: b128 global reads, K-innermost LDS scatter ----
    for (int idx = tid; idx < KT * KD * 8; idx += 256) {   // WW/8 == 8 chunks per row
      int c8 = idx & 7;
      int r  = idx >> 3;         // r = k*KD + kh
      int kh = r % KD;
      int k  = r / KD;
      int yy = y + kh - PAD;
      v8bf v = {};
      if (yy >= 0 && yy < HH)
        v = *(const v8bf*)(act + (((long long)b * IC + ic0 + k) * HH + yy) * WW + c8 * 8);
      #pragma unroll
      for (int j = 0; j < 8; ++j) actlds[kh][c8 * 8 + j + PAD][k] = v[j];
    }
    if (PAD) {
      for (int idx = tid; idx < KT * KD; idx += 256) {
        int kh = idx % KD, k = idx / KD;
        actlds[kh][0][k]      = (__bf16)0.f;
        actlds[kh][LW - 1][k] = (__bf16)0.f;
      }
    }

    for (int kh = 0; kh < KD; ++kh) {
      __syncthreads();     // previous kh's wlds reads complete
      // ---- async-stage weights for taps (kh, kw=0..KD-1): contiguous b128 ----
      for (int idx = tid; idx < KD * (OCH * KT / 8); idx += 256) {
        int kw  = idx >> 10;              // OCH*KT/8 == 1024
        int rem = idx & 1023;
        int o   = rem >> 2;
        int kc  = (rem & 3) << 3;
        const __bf16* src = wgt + (((long long)(kh * KD + kw) * OCH + o) * IC + ic0 + kc);
        unsigned dst = wlds_base + (unsigned)((((kw * OCH + o) * KT) + kc) * 2);
        async_ld_b128(dst, src);
      }
      wait_async0();
      __syncthreads();

      #pragma unroll
      for (int kw = 0; kw < KD; ++kw) {
        // B fragments: lane = pixel column, K = e + 16*lhalf -> one 32B LDS load
        v16bf bfrag[4];
        #pragma unroll
        for (int nt = 0; nt < 4; ++nt) {
          int xcol = nt * 16 + l16 + kw;          // padded column index
          bfrag[nt] = *(const v16bf*)&actlds[kh][xcol][lhalf * 16];
        }
        // A fragments: lane = out channel, K = e + 16*lhalf
        #pragma unroll
        for (int mt = 0; mt < 2; ++mt) {
          int m = m_base + mt * 16 + l16;
          v8bf alo = *(const v8bf*)&wlds[kw][m][lhalf * 16];
          v8bf ahi = *(const v8bf*)&wlds[kw][m][lhalf * 16 + 8];
          v16bf af;
          #pragma unroll
          for (int e = 0; e < 8; ++e) { af[e] = alo[e]; af[e + 8] = ahi[e]; }
          #pragma unroll
          for (int nt = 0; nt < 4; ++nt) {
            acc[mt][nt] = __builtin_amdgcn_wmma_f32_16x16x32_bf16(
                false, af, false, bfrag[nt], (short)0, acc[mt][nt], false, false);
          }
        }
      }
    }
  }

  // Epilogue: D layout -> m = m_base + mt*16 + lhalf*8 + r, x = nt*16 + l16
  #pragma unroll
  for (int mt = 0; mt < 2; ++mt) {
    #pragma unroll
    for (int r = 0; r < 8; ++r) {
      int m = m_base + mt * 16 + lhalf * 8 + r;
      float badd = 0.f, scl = 1.f;
      if (MODE == 0) badd = bias[m] + pervec[b * OCH + m];
      if (MODE == 1) badd = bias[m];
      if (MODE == 2) scl  = pervec[b * OCH + m];
      #pragma unroll
      for (int nt = 0; nt < 4; ++nt) {
        int xx = nt * 16 + l16;
        long long oidx = (((long long)b * OCH + m) * HH + y) * WW + xx;
        float v = acc[mt][nt][r] * scl + badd;
        if (MODE == 2) out[oidx] += v;
        else           out[oidx] = v;
      }
    }
  }
}

// ---------------------------------------------------------------------------
extern "C" void kernel_launch(void* const* d_in, const int* in_sizes, int n_in,
                              void* d_out, int out_size, void* d_ws, size_t ws_size,
                              hipStream_t stream) {
  const float* x       = (const float*)d_in[0];
  const float* temb    = (const float*)d_in[1];
  const float* style   = (const float*)d_in[2];
  const float* gn1_g   = (const float*)d_in[3];
  const float* gn1_b   = (const float*)d_in[4];
  const float* conv1_w = (const float*)d_in[5];
  const float* conv1_b = (const float*)d_in[6];
  const float* temb_w  = (const float*)d_in[7];
  const float* temb_b  = (const float*)d_in[8];
  const float* gn2_g   = (const float*)d_in[9];
  const float* gn2_b   = (const float*)d_in[10];
  const float* style_w = (const float*)d_in[11];
  const float* style_b = (const float*)d_in[12];
  const float* mod_w   = (const float*)d_in[13];
  const float* sc_w    = (const float*)d_in[14];
  const float* sc_b    = (const float*)d_in[15];
  float* out = (float*)d_out;

  char* ws = (char*)d_ws;
  size_t off = 0;
  auto alloc = [&](size_t bytes) -> void* {
    void* p = (void*)(ws + off);
    off += (bytes + 255) & ~(size_t)255;
    return p;
  };

  __bf16* bufA   = (__bf16*)alloc((size_t)NB * ICH * HW * 2);   // x_bf16, then act1
  float*  h1     = (float*) alloc((size_t)NB * OCH * HW * 4);
  __bf16* act2   = (__bf16*)alloc((size_t)NB * OCH * HW * 2);
  __bf16* w1sw   = (__bf16*)alloc((size_t)9 * OCH * ICH * 2);
  __bf16* modwsw = (__bf16*)alloc((size_t)9 * OCH * OCH * 2);
  __bf16* scsw   = (__bf16*)alloc((size_t)OCH * ICH * 2);
  float*  S_o    = (float*) alloc((size_t)OCH * 4);
  float*  tembp  = (float*) alloc((size_t)NB * OCH * 4);
  float*  scale  = (float*) alloc((size_t)NB * OCH * 4);
  float*  stats  = (float*) alloc((size_t)NB * NG * 2 * 4);     // reused GN1/GN2

  const dim3 blk(256);
  const int ewGrid = 2048;

  // --- weight prep ---
  wswizzle_kernel<<<512, blk, 0, stream>>>(conv1_w, w1sw, OCH, ICH, 9);
  wswizzle_kernel<<<1024, blk, 0, stream>>>(mod_w, modwsw, OCH, OCH, 9);
  wswizzle_kernel<<<128, blk, 0, stream>>>(sc_w, scsw, OCH, ICH, 1);
  modw_sumsq_kernel<<<OCH, blk, 0, stream>>>(mod_w, S_o);

  // --- shortcut: x -> bf16, 1x1 conv into d_out ---
  cvt_bf16_kernel<<<ewGrid, blk, 0, stream>>>(x, bufA, (long long)NB * ICH * HW);
  conv_wmma_kernel<ICH, 1, 1><<<NB * HH, blk, 0, stream>>>(bufA, scsw, sc_b, nullptr, out);

  // --- block1: GN -> swish -> conv3x3 (+ temb broadcast) ---
  gn_stats_kernel<ICH><<<NB * NG, blk, 0, stream>>>(x, stats);
  gn_swish_bf16_kernel<ICH><<<ewGrid, blk, 0, stream>>>(x, stats, gn1_g, gn1_b, bufA);
  temb_proj_kernel<<<(NB * OCH + 255) / 256, blk, 0, stream>>>(temb, temb_w, temb_b, tembp);
  conv_wmma_kernel<ICH, 3, 0><<<NB * HH, blk, 0, stream>>>(bufA, w1sw, conv1_b, tembp, h1);

  // --- block2: GN -> swish ---
  gn_stats_kernel<OCH><<<NB * NG, blk, 0, stream>>>(h1, stats);
  gn_swish_bf16_kernel<OCH><<<ewGrid, blk, 0, stream>>>(h1, stats, gn2_g, gn2_b, act2);

  // --- modulated conv: shared weights, per-(b,o) demod scale, accumulate ---
  style_scale_kernel<<<(NB * OCH + 255) / 256, blk, 0, stream>>>(style, style_w, style_b, S_o, scale);
  conv_wmma_kernel<OCH, 3, 2><<<NB * HH, blk, 0, stream>>>(act2, modwsw, nullptr, scale, out);
}
